// CViTFlow_29085518529012
// MI455X (gfx1250) — compile-verified, matrix operand order checked by reading, not run on hardware
//
#include <hip/hip_runtime.h>

// ---------------------------------------------------------------- types
typedef _Float16 v16h __attribute__((ext_vector_type(16)));
typedef _Float16 v8h  __attribute__((ext_vector_type(8)));
typedef float    v8f  __attribute__((ext_vector_type(8)));

// ---------------------------------------------------------------- sizes
#define BB   2
#define TT   2304
#define DIM  256
#define HEADS 8
#define DH   32
#define HW   48
#define NBR  6
#define ROWS (BB*TT)          // 4608 rows per branch

// workspace layout (bytes)
#define WC_OFF  0u
#define WC_BYTES (NBR*DIM*DIM*2u)              // 786432  f16 combined weights [i][o][c]
#define BC_OFF  (WC_OFF + WC_BYTES)            // 786432
#define BC_BYTES (NBR*DIM*4u)                  // 6144    f32 combined bias
#define Y_OFF   (BC_OFF + BC_BYTES)            // 792576  (256B aligned)
#define Y_BYTES (NBR*ROWS*DIM*2u)              // 14155776 f16 dw+bn activations
#define Z_OFF   (Y_OFF + Y_BYTES)              // 14948352
#define Z_BYTES (NBR*ROWS*DIM*2u)              // f16 q/k/v tokens [i][b*t][o]

// -------------------------------------------------- WMMA helpers
__device__ __forceinline__ v8f wmma16(v16h a, v16h b, v8f c) {
    // D = A(16x32 f16) * B(32x16 f16) + C(16x16 f32)
    return __builtin_amdgcn_wmma_f32_16x16x32_f16(false, a, false, b,
                                                  (short)0, c, false, false);
}

// Load one 16-lane-striped operand row: 16 halves at offsets
// {kb..kb+7} and {16+kb..16+kb+7} from `base` (kb = 0 for lanes 0-15, 8 for 16-31).
__device__ __forceinline__ v16h ld_ab(const _Float16* base, int kb) {
    v8h lo = *(const v8h*)(base + kb);
    v8h hi = *(const v8h*)(base + 16 + kb);
    v16h r;
#pragma unroll
    for (int j = 0; j < 8; ++j) { r[j] = lo[j]; r[8 + j] = hi[j]; }
    return r;
}

// =================================================================
// Kernel 1: fold pointwise-conv + linear into one 256x256 f16 matrix
//   Wc[i][o][c] = sum_m lin[i][o][m] * pw[i][m][c];  bc[i][o] = lin[i][o][:]@pw_b[i]
// =================================================================
__global__ void k_prep(const float* __restrict__ pw, const float* __restrict__ pwb,
                       const float* __restrict__ lin,
                       _Float16* __restrict__ Wc, float* __restrict__ bc) {
    int bid = blockIdx.x;                 // i*256 + o
    int c   = threadIdx.x;                // 0..255
    int i   = bid >> 8;
    const float* lrow = lin + (size_t)bid * DIM;        // lin[i][o][*]
    const float* pwi  = pw  + (size_t)i * DIM * DIM;
    float acc = 0.f;
    for (int m = 0; m < DIM; ++m)
        acc += lrow[m] * pwi[(size_t)m * DIM + c];
    Wc[(size_t)bid * DIM + c] = (_Float16)acc;
    if (c == 0) {
        const float* pb = pwb + i * DIM;
        float bb = 0.f;
        for (int m = 0; m < DIM; ++m) bb += lrow[m] * pb[m];
        bc[bid] = bb;
    }
}

// =================================================================
// Kernel 2: depthwise 3x3 (zero pad) + BatchNorm(eval) -> f16
// grid: NBR*BB*TT blocks, 256 threads (one channel each; fully coalesced
// because the token tensor is channel-last)
// =================================================================
__global__ void k_dwbn(const float* __restrict__ x1, const float* __restrict__ x2,
                       const float* __restrict__ dw, const float* __restrict__ gam,
                       const float* __restrict__ bet, const float* __restrict__ mu,
                       const float* __restrict__ var, _Float16* __restrict__ Y) {
    int bid = blockIdx.x;                 // i*(BB*TT) + b*TT + p
    int c   = threadIdx.x;
    int i   = bid / (BB * TT);
    int rem = bid % (BB * TT);
    int b   = rem / TT;
    int p   = rem % TT;
    int hh = p / HW, ww = p % HW;
    const float* x  = (i < 3) ? x1 : x2;
    const float* xb = x + (size_t)b * TT * DIM;
    const float* w  = dw + ((size_t)i * DIM + c) * 9;
    float acc = 0.f;
#pragma unroll
    for (int kh = 0; kh < 3; ++kh) {
        int yy = hh + kh - 1;
        if ((unsigned)yy >= (unsigned)HW) continue;
#pragma unroll
        for (int kw = 0; kw < 3; ++kw) {
            int xx = ww + kw - 1;
            if ((unsigned)xx >= (unsigned)HW) continue;
            acc += w[kh * 3 + kw] * xb[(size_t)(yy * HW + xx) * DIM + c];
        }
    }
    int ic = i * DIM + c;
    float s = gam[ic] * rsqrtf(var[ic] + 1e-5f);
    float o = acc * s + (bet[ic] - mu[ic] * s);
    Y[(size_t)bid * DIM + c] = (_Float16)o;
}

// =================================================================
// Kernel 3: per-branch GEMM  Z = Y (ROWSx256) * Wc^T + bc, f16 out.
// One wave per (branch, 16-row tile); A tiles (16x256) cached in 8 v16h;
// 16 n-tiles x 8 k-chunks = 128 WMMAs per wave.
// =================================================================
__global__ void __launch_bounds__(256) k_gemm(const _Float16* __restrict__ Y,
                                              const _Float16* __restrict__ Wc,
                                              const float* __restrict__ bc,
                                              _Float16* __restrict__ Z) {
    int wv   = blockIdx.x * 8 + (threadIdx.x >> 5);  // 0 .. 6*288-1
    int lane = threadIdx.x & 31;
    int i  = wv / (ROWS / 16);
    int mt = wv % (ROWS / 16);
    int nloc = lane & 15;
    int kb   = (lane < 16) ? 0 : 8;
    int mhi  = (lane < 16) ? 0 : 8;

    const _Float16* arow = Y + ((size_t)i * ROWS + mt * 16 + nloc) * DIM;
    v16h A[8];
#pragma unroll
    for (int kc = 0; kc < 8; ++kc) A[kc] = ld_ab(arow + kc * 32, kb);

    for (int ot = 0; ot < 16; ++ot) {
        int o = ot * 16 + nloc;
        const _Float16* brow = Wc + ((size_t)i * DIM + o) * DIM;
        v8f acc = {};
#pragma unroll
        for (int kc = 0; kc < 8; ++kc) {
            v16h Bt = ld_ab(brow + kc * 32, kb);
            acc = wmma16(A[kc], Bt, acc);
        }
        float bias = bc[i * DIM + o];
#pragma unroll
        for (int r = 0; r < 8; ++r) {
            int row = mt * 16 + r + mhi;
            Z[((size_t)i * ROWS + row) * DIM + o] = (_Float16)(acc[r] + bias);
        }
    }
}

// =================================================================
// Kernel 4: flash softmin cross-attention + residual.
//   logits = -(Q.K^T)/16 ; P = softmax(logits) ; O = P.V + Q
// 8 waves/block, each owns a 16-row Q tile; block shares 32-row K/V LDS tiles.
// =================================================================
__global__ void __launch_bounds__(256) k_attn(const _Float16* __restrict__ Z,
                                              float* __restrict__ out) {
    __shared__ _Float16 Ksh[32][32];       // K tile, row-major [kv][d]
    __shared__ _Float16 VshT[32][32];      // V tile transposed [d][kv]
    __shared__ _Float16 Psh[8][16][32];    // per-wave P bounce [m][kv]

    int bid  = blockIdx.x;
    int side = bid / 288;                  // 0: (q1,k2,v2)  1: (q2,k1,v1)
    int rem  = bid % 288;
    int bh   = rem / 18;
    int qb   = rem % 18;
    int b = bh >> 3, h = bh & 7;
    int tid = threadIdx.x;
    int wave = tid >> 5, lane = tid & 31;
    int nloc = lane & 15;
    int kb   = (lane < 16) ? 0 : 8;
    int mhi  = (lane < 16) ? 0 : 8;

    int qi = side ? 3 : 0, ki = side ? 1 : 4, vi = side ? 2 : 5;
    const _Float16* Qb = Z + ((size_t)qi * ROWS + b * TT) * DIM + h * DH;
    const _Float16* Kb = Z + ((size_t)ki * ROWS + b * TT) * DIM + h * DH;
    const _Float16* Vb = Z + ((size_t)vi * ROWS + b * TT) * DIM + h * DH;

    int q0 = qb * 128 + wave * 16;
    v16h Qa = ld_ab(Qb + (size_t)(q0 + nloc) * DIM, kb);   // Q A-tile (registers)

    float mrun[8], lrun[8];
    v8f acc0 = {}, acc1 = {};
#pragma unroll
    for (int r = 0; r < 8; ++r) { mrun[r] = -1e30f; lrun[r] = 0.f; }

    for (int ch = 0; ch < TT / 32; ++ch) {
        int kv0 = ch * 32;
        {   // cooperative K/V tile load (K as-is, V transposed)
            int u = tid & 127;
            int row = u >> 2, seg = u & 3;
            if (tid < 128) {
                v8h kk = *(const v8h*)(Kb + (size_t)(kv0 + row) * DIM + seg * 8);
                *(v8h*)&Ksh[row][seg * 8] = kk;
                if (ch + 1 < TT / 32)
                    __builtin_prefetch(Kb + (size_t)(kv0 + 32 + row) * DIM + seg * 8, 0, 1);
            } else {
                v8h vv = *(const v8h*)(Vb + (size_t)(kv0 + row) * DIM + seg * 8);
#pragma unroll
                for (int j = 0; j < 8; ++j) VshT[seg * 8 + j][row] = vv[j];
                if (ch + 1 < TT / 32)
                    __builtin_prefetch(Vb + (size_t)(kv0 + 32 + row) * DIM + seg * 8, 0, 1);
            }
        }
        __syncthreads();

        // S = Q.K^T for two 16-col kv sub-tiles
        v16h B0 = ld_ab(&Ksh[nloc][0], kb);
        v16h B1 = ld_ab(&Ksh[16 + nloc][0], kb);
        v8f s0 = {}, s1 = {};
        s0 = wmma16(Qa, B0, s0);
        s1 = wmma16(Qa, B1, s1);

        // online softmax of  -S/16  (softmin); rows live in 16-lane halves
#pragma unroll
        for (int r = 0; r < 8; ++r) {
            float l0 = s0[r] * -0.0625f;
            float l1 = s1[r] * -0.0625f;
            float cm = fmaxf(l0, l1);
            cm = fmaxf(cm, __shfl_xor(cm, 1, 16));
            cm = fmaxf(cm, __shfl_xor(cm, 2, 16));
            cm = fmaxf(cm, __shfl_xor(cm, 4, 16));
            cm = fmaxf(cm, __shfl_xor(cm, 8, 16));
            float nm   = fmaxf(mrun[r], cm);
            float corr = __expf(mrun[r] - nm);
            float p0 = __expf(l0 - nm);
            float p1 = __expf(l1 - nm);
            float rs = p0 + p1;
            rs += __shfl_xor(rs, 1, 16);
            rs += __shfl_xor(rs, 2, 16);
            rs += __shfl_xor(rs, 4, 16);
            rs += __shfl_xor(rs, 8, 16);
            lrun[r] = lrun[r] * corr + rs;
            mrun[r] = nm;
            acc0[r] *= corr;
            acc1[r] *= corr;
            int row = r + mhi;
            Psh[wave][row][nloc]      = (_Float16)p0;   // C-layout -> LDS
            Psh[wave][row][16 + nloc] = (_Float16)p1;
        }
        asm volatile("s_wait_dscnt 0" ::: "memory");    // intra-wave LDS turnaround

        v16h Pa  = ld_ab(&Psh[wave][nloc][0], kb);      // re-read as A operand
        v16h Bv0 = ld_ab(&VshT[nloc][0], kb);
        v16h Bv1 = ld_ab(&VshT[16 + nloc][0], kb);
        acc0 = wmma16(Pa, Bv0, acc0);                   // O += P.V
        acc1 = wmma16(Pa, Bv1, acc1);
        __syncthreads();
    }

    // epilogue: O = acc/l + Q, stored as flat [b,h,t,d] (matches buggy reshape)
    size_t obase = (size_t)side * (BB * HEADS * TT * DH)
                 + ((size_t)(b * HEADS + h) * TT) * DH;
#pragma unroll
    for (int r = 0; r < 8; ++r) {
        int row = r + mhi;
        int t = q0 + row;
        float inv = 1.f / lrun[r];
        float qv0 = (float)Qb[(size_t)t * DIM + nloc];
        float qv1 = (float)Qb[(size_t)t * DIM + 16 + nloc];
        out[obase + (size_t)t * DH + nloc]      = acc0[r] * inv + qv0;
        out[obase + (size_t)t * DH + 16 + nloc] = acc1[r] * inv + qv1;
    }
}

// =================================================================
extern "C" void kernel_launch(void* const* d_in, const int* in_sizes, int n_in,
                              void* d_out, int out_size, void* d_ws, size_t ws_size,
                              hipStream_t stream) {
    (void)in_sizes; (void)n_in; (void)out_size; (void)ws_size;
    const float* x1  = (const float*)d_in[0];
    const float* x2  = (const float*)d_in[1];
    const float* dw  = (const float*)d_in[2];
    const float* gam = (const float*)d_in[3];
    const float* bet = (const float*)d_in[4];
    const float* mu  = (const float*)d_in[5];
    const float* var = (const float*)d_in[6];
    const float* pw  = (const float*)d_in[7];
    const float* pwb = (const float*)d_in[8];
    const float* lin = (const float*)d_in[9];

    char* ws = (char*)d_ws;
    _Float16* Wc = (_Float16*)(ws + WC_OFF);
    float*    bc = (float*)   (ws + BC_OFF);
    _Float16* Y  = (_Float16*)(ws + Y_OFF);
    _Float16* Zq = (_Float16*)(ws + Z_OFF);
    float* out = (float*)d_out;

    k_prep<<<NBR * DIM, DIM, 0, stream>>>(pw, pwb, lin, Wc, bc);
    k_dwbn<<<NBR * BB * TT, DIM, 0, stream>>>(x1, x2, dw, gam, bet, mu, var, Y);
    k_gemm<<<(NBR * (ROWS / 16)) / 8, 256, 0, stream>>>(Y, Wc, bc, Zq);
    k_attn<<<2 * BB * HEADS * (TT / 128), 256, 0, stream>>>(Zq, out);
}
